// MultiHeadAttention_66279935312348
// MI455X (gfx1250) — matrix operand order repaired and checked by reading, not compiled, not run on hardware
//
#include <hip/hip_runtime.h>
#include <stdint.h>
#include <stddef.h>

#define D_MODEL 1024
#define SEQ     4096
#define NHEADS  16
#define DK      64

// ---------------- types ----------------
typedef __bf16 v16bf __attribute__((ext_vector_type(16)));
typedef float  v8f   __attribute__((ext_vector_type(8)));
typedef unsigned int v4u __attribute__((ext_vector_type(4)));

union Frag {           // 32 bytes: one WMMA bf16 A or B operand per lane
    v4u   q[2];
    v16bf v;
};

__device__ inline v8f v8f_zero() {
    v8f v;
#pragma unroll
    for (int i = 0; i < 8; ++i) v[i] = 0.0f;
    return v;
}

__device__ inline v8f wmma_bf16(v16bf a, v16bf b, v8f c) {
    // D = A(16x32 bf16) x B(32x16 bf16) + C(16x16 f32)
    return __builtin_amdgcn_wmma_f32_16x16x32_bf16(
        /*neg_a=*/false, a, /*neg_b=*/false, b,
        /*c_mod=*/(short)0, c, /*reuse_a=*/false, /*reuse_b=*/false);
}

__device__ inline unsigned short f2bf_bits(float f) {
    union { float f; unsigned int u; } c; c.f = f;
    unsigned int u = c.u;
    u += 0x7FFFu + ((u >> 16) & 1u);   // round to nearest even
    return (unsigned short)(u >> 16);
}

// ---------------- fp32 -> bf16 conversion ----------------
__global__ void mha_cvt_bf16(const float* __restrict__ in,
                             unsigned short* __restrict__ out, int n) {
    int i = blockIdx.x * blockDim.x + threadIdx.x;
    if (i < n) out[i] = f2bf_bits(in[i]);
}

// ---------------- GEMM: out = (A[M,K] x W[N,K]^T + bias) * scale ----------------
enum { MODE_BF16_ROW = 0, MODE_BF16_TRANS = 1, MODE_F32_ROW = 2 };

__global__ __launch_bounds__(256)
void mha_gemm_xwT(const unsigned short* __restrict__ A,
                  const unsigned short* __restrict__ W,
                  const float* __restrict__ bias,
                  void* __restrict__ out,
                  int M, int N, int K, float scale, int mode) {
    const int lane = threadIdx.x & 31;
    const int w    = threadIdx.x >> 5;      // 8 waves / block
    const int r    = lane & 15;
    const int hi   = lane >> 4;             // 0 or 1 (half-wave)
    const int wn   = w & 1;                 // 2 waves over N (64 each)
    const int wm   = w >> 1;                // 4 waves over M (32 each)
    const int M0   = blockIdx.y * 128 + wm * 32;
    const int N0   = blockIdx.x * 128 + wn * 64;

    v8f acc[2][4];
#pragma unroll
    for (int im = 0; im < 2; ++im)
#pragma unroll
        for (int in = 0; in < 4; ++in) acc[im][in] = v8f_zero();

    for (int k0 = 0; k0 < K; k0 += 32) {
        Frag a[2], b[4];
#pragma unroll
        for (int im = 0; im < 2; ++im) {
            const unsigned short* p = A + (size_t)(M0 + im * 16 + r) * K + k0 + hi * 8;
            a[im].q[0] = *reinterpret_cast<const v4u*>(p);
            a[im].q[1] = *reinterpret_cast<const v4u*>(p + 16);
        }
#pragma unroll
        for (int in = 0; in < 4; ++in) {
            const unsigned short* p = W + (size_t)(N0 + in * 16 + r) * K + k0 + hi * 8;
            b[in].q[0] = *reinterpret_cast<const v4u*>(p);
            b[in].q[1] = *reinterpret_cast<const v4u*>(p + 16);
        }
#pragma unroll
        for (int im = 0; im < 2; ++im)
#pragma unroll
            for (int in = 0; in < 4; ++in)
                acc[im][in] = wmma_bf16(a[im].v, b[in].v, acc[im][in]);
    }

#pragma unroll
    for (int in = 0; in < 4; ++in) {
        const int col = N0 + in * 16 + r;
        const float bv = bias[col];
#pragma unroll
        for (int im = 0; im < 2; ++im) {
#pragma unroll
            for (int i = 0; i < 8; ++i) {
                const int row = M0 + im * 16 + i + hi * 8;  // C layout (§7.12.2)
                const float val = (acc[im][in][i] + bv) * scale;
                if (mode == MODE_F32_ROW)
                    ((float*)out)[(size_t)row * N + col] = val;
                else if (mode == MODE_BF16_ROW)
                    ((unsigned short*)out)[(size_t)row * N + col] = f2bf_bits(val);
                else // MODE_BF16_TRANS: out[N][M]
                    ((unsigned short*)out)[(size_t)col * M + row] = f2bf_bits(val);
            }
        }
    }
}

// ---------------- Flash attention ----------------
// Workgroup = (head, 256 query rows); 8 waves x 32 rows each.
// K/V blocks staged in LDS with async global->LDS copies, double buffered.
#define K_STRIDE 72   // halves per K row in LDS (64 + pad, 16B-aligned)
#define V_STRIDE 40   // halves per V row in LDS (32 + pad, 16B-aligned)
#define NBLK     (SEQ / 32)

__device__ inline void async_copy_kv(const unsigned short* __restrict__ Km,
                                     const unsigned short* __restrict__ Vt,
                                     unsigned short* kdst, unsigned short* vdst,
                                     int n0, int cq, int tid) {
    // K block: 32 rows x 64 halves (128B/row -> 8 threads per row, 16B each)
    {
        const int row = tid >> 3, chunk = tid & 7;
        const unsigned lds = (unsigned)(uintptr_t)(kdst + row * K_STRIDE + chunk * 8);
        const unsigned short* g = Km + (size_t)(n0 + row) * D_MODEL + cq + chunk * 8;
        asm volatile("global_load_async_to_lds_b128 %0, %1, off"
                     :: "v"(lds), "v"(g) : "memory");
    }
    // V block: 64 rows x 32 halves (64B/row -> 4 threads per row, 16B each)
    {
        const int row = tid >> 2, chunk = tid & 3;
        const unsigned lds = (unsigned)(uintptr_t)(vdst + row * V_STRIDE + chunk * 8);
        const unsigned short* g = Vt + (size_t)(cq + row) * SEQ + n0 + chunk * 8;
        asm volatile("global_load_async_to_lds_b128 %0, %1, off"
                     :: "v"(lds), "v"(g) : "memory");
    }
}

__global__ __launch_bounds__(256)
void mha_flash_attn(const unsigned short* __restrict__ Q,
                    const unsigned short* __restrict__ Km,
                    const unsigned short* __restrict__ Vt,
                    const int* __restrict__ mask,
                    unsigned short* __restrict__ O) {
    __shared__ __align__(16) unsigned short Kbuf[2][32 * K_STRIDE]; // 9 KB
    __shared__ __align__(16) unsigned short Vbuf[2][64 * V_STRIDE]; // 10 KB
    __shared__ __align__(16) unsigned short Pbuf[8][2][16 * 40];    // 20 KB

    const int tid  = threadIdx.x;
    const int lane = tid & 31;
    const int w    = tid >> 5;
    const int h    = blockIdx.y;            // head
    const int qb   = blockIdx.x;            // 256-row query block
    const int m0   = qb * 256 + w * 32;     // wave's first query row
    const int r    = lane & 15;
    const int hi   = lane >> 4;
    const int cq   = h * DK;

    // Q A-fragments: [m-tile][k-step]
    Frag aq[2][2];
#pragma unroll
    for (int mt = 0; mt < 2; ++mt)
#pragma unroll
        for (int kk = 0; kk < 2; ++kk) {
            const unsigned short* p =
                Q + (size_t)(m0 + mt * 16 + r) * D_MODEL + cq + kk * 32 + hi * 8;
            aq[mt][kk].q[0] = *reinterpret_cast<const v4u*>(p);
            aq[mt][kk].q[1] = *reinterpret_cast<const v4u*>(p + 16);
        }

    v8f o[2][4];
    float mrow[2][8], lrow[2][8];
#pragma unroll
    for (int mt = 0; mt < 2; ++mt) {
#pragma unroll
        for (int t = 0; t < 4; ++t) o[mt][t] = v8f_zero();
#pragma unroll
        for (int i = 0; i < 8; ++i) { mrow[mt][i] = -1e30f; lrow[mt][i] = 0.0f; }
    }

    // prologue: prefetch block 0
    async_copy_kv(Km, Vt, Kbuf[0], Vbuf[0], 0, cq, tid);

    for (int blk = 0; blk < NBLK; ++blk) {
        const int cur = blk & 1;
        const int n0  = blk * 32;

        if (blk + 1 < NBLK) {
            async_copy_kv(Km, Vt, Kbuf[cur ^ 1], Vbuf[cur ^ 1], n0 + 32, cq, tid);
            asm volatile("s_wait_asynccnt 0x2" ::: "memory"); // cur block landed
        } else {
            asm volatile("s_wait_asynccnt 0x0" ::: "memory");
        }
        __syncthreads();  // all waves' copies visible

        const float madd0 = mask[n0 + r]      ? 0.0f : -1e30f;
        const float madd1 = mask[n0 + 16 + r] ? 0.0f : -1e30f;

        // K B-fragments from LDS (shared by both m-tiles)
        Frag bk[2][2];
#pragma unroll
        for (int nh = 0; nh < 2; ++nh)
#pragma unroll
            for (int rs = 0; rs < 2; ++rs) {
                const unsigned short* p =
                    &Kbuf[cur][(nh * 16 + r) * K_STRIDE + rs * 32 + hi * 8];
                bk[nh][rs].q[0] = *reinterpret_cast<const v4u*>(p);
                bk[nh][rs].q[1] = *reinterpret_cast<const v4u*>(p + 16);
            }

        Frag ap[2];
#pragma unroll
        for (int mt = 0; mt < 2; ++mt) {
            v8f s0 = v8f_zero(), s1 = v8f_zero();
            s0 = wmma_bf16(aq[mt][0].v, bk[0][0].v, s0);
            s0 = wmma_bf16(aq[mt][1].v, bk[0][1].v, s0);
            s1 = wmma_bf16(aq[mt][0].v, bk[1][0].v, s1);
            s1 = wmma_bf16(aq[mt][1].v, bk[1][1].v, s1);

#pragma unroll
            for (int i = 0; i < 8; ++i) { s0[i] += madd0; s1[i] += madd1; }

            float p0[8], p1[8];
#pragma unroll
            for (int i = 0; i < 8; ++i) {
                float bm = fmaxf(s0[i], s1[i]);
#pragma unroll
                for (int off = 1; off < 16; off <<= 1)
                    bm = fmaxf(bm, __shfl_xor(bm, off, 32));
                const float mnew  = fmaxf(mrow[mt][i], bm);
                const float alpha = __expf(mrow[mt][i] - mnew);
                p0[i] = __expf(s0[i] - mnew);
                p1[i] = __expf(s1[i] - mnew);
                float ps = p0[i] + p1[i];
#pragma unroll
                for (int off = 1; off < 16; off <<= 1)
                    ps += __shfl_xor(ps, off, 32);
                lrow[mt][i] = lrow[mt][i] * alpha + ps;
                mrow[mt][i] = mnew;
#pragma unroll
                for (int t = 0; t < 4; ++t) o[mt][t][i] *= alpha;
            }

            // re-layout P: C-layout regs -> LDS -> A-operand fragment
            unsigned short* pb = Pbuf[w][mt];
#pragma unroll
            for (int i = 0; i < 8; ++i) {
                const int prow = i + hi * 8;
                pb[prow * 40 + r]      = f2bf_bits(p0[i]);
                pb[prow * 40 + 16 + r] = f2bf_bits(p1[i]);
            }
            __builtin_amdgcn_wave_barrier();  // LDS in-order per wave

            const unsigned short* p = pb + r * 40 + hi * 8;
            ap[mt].q[0] = *reinterpret_cast<const v4u*>(p);
            ap[mt].q[1] = *reinterpret_cast<const v4u*>(p + 16);
        }

        // O += P(16x32) x V(32x64); each V fragment feeds both m-tiles
#pragma unroll
        for (int t = 0; t < 4; ++t) {
            Frag bvv;
            const unsigned short* p = &Vbuf[cur][(t * 16 + r) * V_STRIDE + hi * 8];
            bvv.q[0] = *reinterpret_cast<const v4u*>(p);
            bvv.q[1] = *reinterpret_cast<const v4u*>(p + 16);
            o[0][t] = wmma_bf16(ap[0].v, bvv.v, o[0][t]);
            o[1][t] = wmma_bf16(ap[1].v, bvv.v, o[1][t]);
        }

        __syncthreads();  // everyone done reading cur before it is overwritten
    }

    // normalize and store attention output (bf16, [SEQ][D_MODEL])
#pragma unroll
    for (int mt = 0; mt < 2; ++mt)
#pragma unroll
        for (int i = 0; i < 8; ++i) {
            const float inv = 1.0f / lrow[mt][i];
            const int row = m0 + mt * 16 + i + hi * 8;
#pragma unroll
            for (int t = 0; t < 4; ++t)
                O[(size_t)row * D_MODEL + cq + t * 16 + r] =
                    f2bf_bits(o[mt][t][i] * inv);
        }
}

// ---------------- host-side orchestration ----------------
extern "C" void kernel_launch(void* const* d_in, const int* in_sizes, int n_in,
                              void* d_out, int out_size, void* d_ws, size_t ws_size,
                              hipStream_t stream) {
    const float* x   = (const float*)d_in[0];
    const int*  msk  = (const int*)  d_in[1];
    const float* Wq  = (const float*)d_in[2];
    const float* bq  = (const float*)d_in[3];
    const float* Wk  = (const float*)d_in[4];
    const float* bk  = (const float*)d_in[5];
    const float* Wv  = (const float*)d_in[6];
    const float* bv  = (const float*)d_in[7];
    const float* Wo  = (const float*)d_in[8];
    const float* bo  = (const float*)d_in[9];

    char* ws = (char*)d_ws;
    const size_t SZ_X = (size_t)SEQ * D_MODEL * sizeof(unsigned short);     // 8 MB
    const size_t SZ_W = (size_t)D_MODEL * D_MODEL * sizeof(unsigned short); // 2 MB
    unsigned short* xb  = (unsigned short*)(ws);
    unsigned short* Wqb = (unsigned short*)(ws + SZ_X);
    unsigned short* Wkb = (unsigned short*)(ws + SZ_X + 1 * SZ_W);
    unsigned short* Wvb = (unsigned short*)(ws + SZ_X + 2 * SZ_W);
    unsigned short* Wob = (unsigned short*)(ws + SZ_X + 3 * SZ_W);
    unsigned short* Qb  = (unsigned short*)(ws + SZ_X + 4 * SZ_W);
    unsigned short* Kb  = Qb  + (size_t)SEQ * D_MODEL;
    unsigned short* Vtb = Kb  + (size_t)SEQ * D_MODEL;
    unsigned short* Ab  = Vtb + (size_t)SEQ * D_MODEL;

    const int nX = SEQ * D_MODEL;
    const int nW = D_MODEL * D_MODEL;
    mha_cvt_bf16<<<(nX + 255) / 256, 256, 0, stream>>>(x,  xb,  nX);
    mha_cvt_bf16<<<(nW + 255) / 256, 256, 0, stream>>>(Wq, Wqb, nW);
    mha_cvt_bf16<<<(nW + 255) / 256, 256, 0, stream>>>(Wk, Wkb, nW);
    mha_cvt_bf16<<<(nW + 255) / 256, 256, 0, stream>>>(Wv, Wvb, nW);
    mha_cvt_bf16<<<(nW + 255) / 256, 256, 0, stream>>>(Wo, Wob, nW);

    dim3 gg(D_MODEL / 128, SEQ / 128);  // (8, 32), 256 threads/block
    // Q pre-scaled by 1/sqrt(dk) = 0.125 so scores need no extra scaling
    mha_gemm_xwT<<<gg, 256, 0, stream>>>(xb, Wqb, bq, Qb, SEQ, D_MODEL, D_MODEL,
                                         0.125f, MODE_BF16_ROW);
    mha_gemm_xwT<<<gg, 256, 0, stream>>>(xb, Wkb, bk, Kb, SEQ, D_MODEL, D_MODEL,
                                         1.0f, MODE_BF16_ROW);
    mha_gemm_xwT<<<gg, 256, 0, stream>>>(xb, Wvb, bv, Vtb, SEQ, D_MODEL, D_MODEL,
                                         1.0f, MODE_BF16_TRANS);

    dim3 ga(SEQ / 256, NHEADS);         // (16, 16) workgroups
    mha_flash_attn<<<ga, 256, 0, stream>>>(Qb, Kb, Vtb, msk, Ab);

    mha_gemm_xwT<<<gg, 256, 0, stream>>>(Ab, Wob, bo, d_out, SEQ, D_MODEL, D_MODEL,
                                         1.0f, MODE_F32_ROW);
}
